// CortexModel_77360950935933
// MI455X (gfx1250) — compile-verified, matrix operand order checked by reading, not run on hardware
//
#include <hip/hip_runtime.h>

// ---- model constants (match reference) ----
#define D_DIM 256
#define NHEAD 8
#define DHEAD 32
#define NLAT  64
#define NB    16
#define EMAXD 13568

typedef __bf16 bf16;
typedef bf16  v16bf __attribute__((ext_vector_type(16)));
typedef float v8f   __attribute__((ext_vector_type(8)));

union AFrag { v16bf v; unsigned short h[16]; };

// Optional gfx1250 async global->LDS DMA path (ASYNCcnt). Guarded so the build
// never breaks if this toolchain doesn't declare the builtins.
#if __has_builtin(__builtin_amdgcn_global_load_async_to_lds_b128) && \
    __has_builtin(__builtin_amdgcn_s_wait_asynccnt)
#define USE_ASYNC_LDS 1
typedef int v4i_t __attribute__((vector_size(16)));
typedef __attribute__((address_space(1))) v4i_t* as1_v4i;
typedef __attribute__((address_space(3))) v4i_t* as3_v4i;
#else
#define USE_ASYNC_LDS 0
#endif

__device__ __forceinline__ unsigned short f2bf(float f) {
  union { float f; unsigned u; } x; x.f = f;
  unsigned u = x.u;
  unsigned r = (u + 0x7fffu + ((u >> 16) & 1u)) >> 16;  // round-nearest-even
  return (unsigned short)r;
}
__device__ __forceinline__ v8f zero8() {
  v8f z;
#pragma unroll
  for (int i = 0; i < 8; ++i) z[i] = 0.f;
  return z;
}
__device__ __forceinline__ v8f wmma_bf16(const AFrag& a, const AFrag& b, v8f c) {
  return __builtin_amdgcn_wmma_f32_16x16x32_bf16(false, a.v, false, b.v, (short)0, c, false, false);
}
__device__ __forceinline__ float gelu_tanh(float x) {
  float x3 = x * x * x;
  return 0.5f * x * (1.0f + tanhf(0.7978845608028654f * (x + 0.044715f * x3)));
}

// ---------------- small utility kernels ----------------
__global__ void cvt_kernel(const float* __restrict__ x, unsigned short* __restrict__ y, int n) {
  int i = blockIdx.x * blockDim.x + threadIdx.x;
  if (i < n) y[i] = f2bf(x[i]);
}

__global__ void pos_kernel(const int* __restrict__ bidx, int* __restrict__ counts,
                           int* __restrict__ pos, int n) {
  int i = blockIdx.x * blockDim.x + threadIdx.x;
  if (i < n) pos[i] = atomicAdd(&counts[bidx[i]], 1);
}

__global__ void lat_init_kernel(const float* __restrict__ li, float* __restrict__ lat, int n) {
  int i = blockIdx.x * blockDim.x + threadIdx.x;
  if (i < n) lat[i] = li[i & (NLAT * D_DIM - 1)];
}

__global__ __launch_bounds__(256) void ln_kernel(const float* __restrict__ X,
                                                 unsigned short* __restrict__ Y,
                                                 const float* __restrict__ g,
                                                 const float* __restrict__ bta) {
  __shared__ float red[256];
  const int r = blockIdx.x, t = threadIdx.x;
  float v = X[(size_t)r * D_DIM + t];
  red[t] = v; __syncthreads();
  for (int s = 128; s > 0; s >>= 1) { if (t < s) red[t] += red[t + s]; __syncthreads(); }
  float mean = red[0] * (1.f / D_DIM);
  __syncthreads();
  float d = v - mean;
  red[t] = d * d; __syncthreads();
  for (int s = 128; s > 0; s >>= 1) { if (t < s) red[t] += red[t + s]; __syncthreads(); }
  float var = red[0] * (1.f / D_DIM);
  float y = d * rsqrtf(var + 1e-5f);
  if (g) y = y * g[t] + bta[t];
  Y[(size_t)r * D_DIM + t] = f2bf(y);
}

__global__ void mean_kernel(const float* __restrict__ lat, unsigned short* __restrict__ out) {
  int b = blockIdx.x, c = threadIdx.x;
  float s = 0.f;
  for (int l = 0; l < NLAT; ++l) s += lat[((size_t)b * NLAT + l) * D_DIM + c];
  out[b * D_DIM + c] = f2bf(s * (1.f / NLAT));
}

__global__ __launch_bounds__(256) void bdot_kernel(const float* __restrict__ bo,
                                                   const float* __restrict__ wo,
                                                   float* __restrict__ out) {
  __shared__ float red[256];
  int r = blockIdx.x, t = threadIdx.x;
  red[t] = bo[(size_t)r * D_DIM + t] * wo[t];
  __syncthreads();
  for (int s = 128; s > 0; s >>= 1) { if (t < s) red[t] += red[t + s]; __syncthreads(); }
  if (t == 0) out[r] = red[0];
}

// ---------------- fused tokenize + LN + KV projection ----------------
// 16 events/block; gathers 3 embeddings (b128), LayerNorm, (16x256)@(256x512) WMMA,
// result transposed through LDS and written as contiguous b128 rows at (batch,pos).
__global__ __launch_bounds__(256) void tok_kv_kernel(
    const int* __restrict__ nid, const int* __restrict__ tbin, const int* __restrict__ val,
    const int* __restrict__ bidx, const int* __restrict__ pos,
    const float* __restrict__ nemb, const float* __restrict__ temb, const float* __restrict__ vemb,
    const unsigned short* __restrict__ wkv, unsigned short* __restrict__ kvout, int nE) {
  __shared__ __align__(32) unsigned short xt[16][D_DIM];       // LN'd tokens, bf16
  __shared__ __align__(32) unsigned short wT[2 * D_DIM][32];   // W chunk, transposed
  __shared__ __align__(32) unsigned short out_lds[16][512];    // K|V rows, bf16
  __shared__ float r1[16][16], r2[16][16];
  __shared__ float smean[16], srstd[16];
  __shared__ int ev_b[16], ev_p[16];

  const int t = threadIdx.x;
  const int row = t >> 4, seg = t & 15;
  const int e = blockIdx.x * 16 + row;

  float v[16];
  if (e < nE) {
    const float4* pn = (const float4*)(nemb + (size_t)nid[e]  * D_DIM + seg * 16);
    const float4* pt = (const float4*)(temb + (size_t)tbin[e] * D_DIM + seg * 16);
    const float4* pv = (const float4*)(vemb + (size_t)val[e]  * D_DIM + seg * 16);
    float s1 = 0.f, s2 = 0.f;
#pragma unroll
    for (int q = 0; q < 4; ++q) {
      float4 a = pn[q], bq4 = pt[q], c4 = pv[q];
      float x0 = a.x + bq4.x + c4.x, x1 = a.y + bq4.y + c4.y;
      float x2 = a.z + bq4.z + c4.z, x3 = a.w + bq4.w + c4.w;
      v[q * 4 + 0] = x0; v[q * 4 + 1] = x1; v[q * 4 + 2] = x2; v[q * 4 + 3] = x3;
      s1 += x0 + x1 + x2 + x3;
      s2 += x0 * x0 + x1 * x1 + x2 * x2 + x3 * x3;
    }
    r1[row][seg] = s1; r2[row][seg] = s2;
    if (seg == 0) { ev_b[row] = bidx[e]; ev_p[row] = pos[e]; }
  } else {
#pragma unroll
    for (int j = 0; j < 16; ++j) v[j] = 0.f;
    r1[row][seg] = 0.f; r2[row][seg] = 0.f;
    if (seg == 0) { ev_b[row] = -1; ev_p[row] = 0; }
  }
  __syncthreads();
  if (seg == 0) {
    float a = 0.f, bq = 0.f;
    for (int j = 0; j < 16; ++j) { a += r1[row][j]; bq += r2[row][j]; }
    float mean = a * (1.f / D_DIM);
    float var  = bq * (1.f / D_DIM) - mean * mean;
    smean[row] = mean; srstd[row] = rsqrtf(var + 1e-5f);
  }
  __syncthreads();
  {
    float mean = smean[row], rs = srstd[row];
#pragma unroll
    for (int j = 0; j < 16; ++j) xt[row][seg * 16 + j] = f2bf((v[j] - mean) * rs);
  }
  __syncthreads();

  const int wave = t >> 5, lane = t & 31, hlf = lane >> 4, l16 = lane & 15;
  v8f acc[4];
#pragma unroll
  for (int c = 0; c < 4; ++c) acc[c] = zero8();

  const int kk0 = t >> 3;         // 0..31  (k row within chunk)
  const int cc0 = (t & 7) * 64;   // column slice base
  for (int kc = 0; kc < 8; ++kc) {
    const int k0 = kc * 32;
    // stage W chunk (32 x 512) transposed into LDS
    {
      const unsigned short* gp = wkv + (size_t)(k0 + kk0) * 512 + cc0;
      __builtin_prefetch(wkv + (size_t)(k0 + 32 + kk0) * 512 + cc0, 0, 0);
#pragma unroll
      for (int jq = 0; jq < 8; ++jq) {
        union { uint4 u; unsigned short s[8]; } w;
        w.u = *(const uint4*)(gp + jq * 8);
#pragma unroll
        for (int j = 0; j < 8; ++j) wT[cc0 + jq * 8 + j][kk0] = w.s[j];
      }
    }
    __syncthreads();
    AFrag a; a.v = *(const v16bf*)&xt[l16][k0 + hlf * 16];
#pragma unroll
    for (int c = 0; c < 4; ++c) {
      const int col0 = (wave * 4 + c) * 16;
      AFrag bfr; bfr.v = *(const v16bf*)&wT[col0 + l16][hlf * 16];
      acc[c] = wmma_bf16(a, bfr, acc[c]);
    }
    __syncthreads();
  }
  // transpose accumulators through LDS, then wide contiguous global stores
#pragma unroll
  for (int c = 0; c < 4; ++c) {
    const int col = (wave * 4 + c) * 16 + l16;
#pragma unroll
    for (int i = 0; i < 8; ++i) out_lds[i + 8 * hlf][col] = f2bf(acc[c][i]);
  }
  __syncthreads();
  {
    const int bb = ev_b[row];
    if (bb >= 0) {
      const unsigned short* src = &out_lds[row][seg * 32];
      unsigned short* dst = kvout + ((size_t)bb * EMAXD + (size_t)ev_p[row]) * 512 + seg * 32;
      uint4 u0 = *(const uint4*)(src);
      uint4 u1 = *(const uint4*)(src + 8);
      uint4 u2 = *(const uint4*)(src + 16);
      uint4 u3 = *(const uint4*)(src + 24);
      *(uint4*)(dst)      = u0;
      *(uint4*)(dst + 8)  = u1;
      *(uint4*)(dst + 16) = u2;
      *(uint4*)(dst + 24) = u3;
    }
  }
}

// ---------------- generic flash attention (bf16 in, online softmax, WMMA) --------
// grid = (ceil(rows_q/64), H, B), block = 128 (4 waves, one 16-row q tile each).
// Out-of-range keys are masked in the score domain, so K/V staging may fetch a
// clamped valid row -> K staging can be a pure async DMA (no zero fill needed).
__global__ __launch_bounds__(128) void flash_kernel(
    const unsigned short* __restrict__ qb, long long q_bstr, int q_rstr,
    const unsigned short* __restrict__ kvb, long long kv_bstr, int kv_rstr,
    int k_coff, int v_coff,
    const int* __restrict__ seqlens, int seq_const,
    void* __restrict__ outp, int out_f32, long long o_bstr, int o_rstr,
    int rows_q, float scale) {
  __shared__ __align__(32) unsigned short k_lds[32][32];     // [key][dh]
  __shared__ __align__(32) unsigned short v_lds[32][32];     // transposed: [dh][key]
  __shared__ __align__(32) unsigned short p_lds[4][16][32];  // per-wave P

  const int b = blockIdx.z, h = blockIdx.y, qt = blockIdx.x;
  const int t = threadIdx.x;
  const int wave = t >> 5, lane = t & 31, hlf = lane >> 4, l16 = lane & 15;
  const int seqlen = seqlens ? seqlens[b] : seq_const;

  AFrag qa;
  {
    const int r = qt * 64 + wave * 16 + l16;
    if (r < rows_q)
      qa.v = *(const v16bf*)(qb + (size_t)b * q_bstr + (size_t)r * q_rstr + h * DHEAD + hlf * 16);
    else
      for (int j = 0; j < 16; ++j) qa.h[j] = 0;
  }

  float m[8], lsum[8];
  v8f o0 = zero8(), o1 = zero8();
#pragma unroll
  for (int i = 0; i < 8; ++i) { m[i] = -1e30f; lsum[i] = 0.f; }

  const int skey = t & 31, spart = t >> 5;  // staging: 32 keys x 4 parts x 8 dh
  const int nch = (seqlen + 31) >> 5;
  for (int ch = 0; ch < nch; ++ch) {
    const int key0 = ch << 5;
    {
      int gk = key0 + skey;
      if (gk >= seqlen) gk = seqlen - 1;  // clamp: contents masked later
      const unsigned short* kp = kvb + (size_t)b * kv_bstr + (size_t)gk * kv_rstr;
      __builtin_prefetch(kp + (size_t)32 * kv_rstr + k_coff + h * DHEAD, 0, 0);
#if USE_ASYNC_LDS
      __builtin_amdgcn_global_load_async_to_lds_b128(
          (as1_v4i)(kp + k_coff + h * DHEAD + spart * 8),
          (as3_v4i)&k_lds[skey][spart * 8], 0, 0);
#else
      *(uint4*)&k_lds[skey][spart * 8] = *(const uint4*)(kp + k_coff + h * DHEAD + spart * 8);
#endif
      union { uint4 u; unsigned short s[8]; } wv;
      wv.u = *(const uint4*)(kp + v_coff + h * DHEAD + spart * 8);
#pragma unroll
      for (int j = 0; j < 8; ++j) v_lds[spart * 8 + j][skey] = wv.s[j];
#if USE_ASYNC_LDS
      __builtin_amdgcn_s_wait_asynccnt(0);
#endif
    }
    __syncthreads();

    // S = Q K^T (two 16-key tiles per chunk), masked + scaled
    float s0[8], s1[8];
    {
      AFrag kb0, kb1;
      kb0.v = *(const v16bf*)&k_lds[l16][hlf * 16];
      kb1.v = *(const v16bf*)&k_lds[16 + l16][hlf * 16];
      v8f c0 = wmma_bf16(qa, kb0, zero8());
      v8f c1 = wmma_bf16(qa, kb1, zero8());
      const int ka = key0 + l16, kb2 = key0 + 16 + l16;
#pragma unroll
      for (int i = 0; i < 8; ++i) {
        s0[i] = (ka  < seqlen) ? c0[i] * scale : -1e30f;
        s1[i] = (kb2 < seqlen) ? c1[i] * scale : -1e30f;
      }
    }
    // online softmax per row (rows live per 16-lane half; reduce with shuffles)
#pragma unroll
    for (int i = 0; i < 8; ++i) {
      float mi = fmaxf(s0[i], s1[i]);
      for (int d = 1; d < 16; d <<= 1) mi = fmaxf(mi, __shfl_xor(mi, d, 32));
      float mn = fmaxf(m[i], mi);
      float alpha = __expf(m[i] - mn);
      float p0 = __expf(s0[i] - mn);
      float p1 = __expf(s1[i] - mn);
      float ps = p0 + p1;
      for (int d = 1; d < 16; d <<= 1) ps += __shfl_xor(ps, d, 32);
      m[i] = mn;
      lsum[i] = lsum[i] * alpha + ps;
      o0[i] *= alpha; o1[i] *= alpha;
      p_lds[wave][i + 8 * hlf][l16]      = f2bf(p0);
      p_lds[wave][i + 8 * hlf][16 + l16] = f2bf(p1);
    }
    // O += P @ V
    {
      AFrag pa, vb0, vb1;
      pa.v  = *(const v16bf*)&p_lds[wave][l16][hlf * 16];
      vb0.v = *(const v16bf*)&v_lds[l16][hlf * 16];
      vb1.v = *(const v16bf*)&v_lds[16 + l16][hlf * 16];
      o0 = wmma_bf16(pa, vb0, o0);
      o1 = wmma_bf16(pa, vb1, o1);
    }
    __syncthreads();
  }
  // epilogue: normalize and store
#pragma unroll
  for (int i = 0; i < 8; ++i) {
    const int gr = qt * 64 + wave * 16 + i + 8 * hlf;
    if (gr < rows_q) {
      const float inv = 1.f / lsum[i];
      const size_t base = (size_t)b * o_bstr + (size_t)gr * o_rstr + h * DHEAD;
      const float w0 = o0[i] * inv, w1 = o1[i] * inv;
      if (out_f32) {
        ((float*)outp)[base + l16] = w0;
        ((float*)outp)[base + 16 + l16] = w1;
      } else {
        ((unsigned short*)outp)[base + l16] = f2bf(w0);
        ((unsigned short*)outp)[base + 16 + l16] = f2bf(w1);
      }
    }
  }
}

// ---------------- generic bf16 WMMA GEMM with fused epilogue ----------------
#define GEMM_OUT_BF16 1
#define GEMM_GELU 2
// grid = (ceil(M/16), N/64), block = 128
__global__ __launch_bounds__(128) void gemm_kernel(
    const unsigned short* __restrict__ X, int M, int K, int N,
    const unsigned short* __restrict__ W,
    const float* __restrict__ bias, const float* __restrict__ resid,
    void* __restrict__ outp, int flags) {
  __shared__ __align__(32) unsigned short wT[64][32];  // W chunk transposed
  const int t = threadIdx.x;
  const int wave = t >> 5, lane = t & 31, hlf = lane >> 4, l16 = lane & 15;
  const int r0 = blockIdx.x * 16;
  const int cblk = blockIdx.y * 64;

  v8f acc = zero8();
  const int kk = t >> 2;         // 0..31
  const int cc = (t & 3) * 16;   // 0,16,32,48
  const int nch = K >> 5;
  for (int kc = 0; kc < nch; ++kc) {
    const int k0 = kc * 32;
    {
      const unsigned short* gp = W + (size_t)(k0 + kk) * N + cblk + cc;
      union { uint4 u; unsigned short s[8]; } w0, w1;
      w0.u = *(const uint4*)gp;
      w1.u = *(const uint4*)(gp + 8);
#pragma unroll
      for (int j = 0; j < 8; ++j) { wT[cc + j][kk] = w0.s[j]; wT[cc + 8 + j][kk] = w1.s[j]; }
    }
    __syncthreads();
    AFrag a;
    {
      const int r = r0 + l16;
      if (r < M) a.v = *(const v16bf*)(X + (size_t)r * K + k0 + hlf * 16);
      else for (int j = 0; j < 16; ++j) a.h[j] = 0;
    }
    AFrag bfr; bfr.v = *(const v16bf*)&wT[wave * 16 + l16][hlf * 16];
    acc = wmma_bf16(a, bfr, acc);
    __syncthreads();
  }
  const int col = cblk + wave * 16 + l16;
#pragma unroll
  for (int i = 0; i < 8; ++i) {
    const int r = r0 + i + 8 * hlf;
    if (r < M) {
      float vv = acc[i];
      if (bias) vv += bias[col];
      if (flags & GEMM_GELU) vv = gelu_tanh(vv);
      const size_t o = (size_t)r * N + col;
      if (flags & GEMM_OUT_BF16) ((unsigned short*)outp)[o] = f2bf(vv);
      else {
        const float rr = resid ? resid[o] : 0.f;
        ((float*)outp)[o] = vv + rr;
      }
    }
  }
}

// ---------------- host orchestration ----------------
extern "C" void kernel_launch(void* const* d_in, const int* in_sizes, int n_in,
                              void* d_out, int out_size, void* d_ws, size_t ws_size,
                              hipStream_t stream) {
  (void)n_in; (void)out_size; (void)ws_size;
  const int*   neuron_ids = (const int*)d_in[0];
  const int*   time_bins  = (const int*)d_in[1];
  const int*   values     = (const int*)d_in[2];
  const int*   bidx       = (const int*)d_in[3];
  const float* nemb       = (const float*)d_in[6];
  const float* temb       = (const float*)d_in[7];
  const float* vemb       = (const float*)d_in[8];
  const float* lat_i      = (const float*)d_in[9];
  const float* Wq_c  = (const float*)d_in[10];
  const float* Wkv_c = (const float*)d_in[11];
  const float* Wo_c  = (const float*)d_in[12];
  const float* W1_c  = (const float*)d_in[13];
  const float* W2_c  = (const float*)d_in[14];
  const float* Wqkv_s = (const float*)d_in[15];
  const float* Wo_s   = (const float*)d_in[16];
  const float* W1_s   = (const float*)d_in[17];
  const float* W2_s   = (const float*)d_in[18];
  const float* bh_query = (const float*)d_in[19];
  const float* bh_wq    = (const float*)d_in[20];
  const float* bh_wkv   = (const float*)d_in[21];
  const float* bh_wo    = (const float*)d_in[22];
  const float* bh_ln_w  = (const float*)d_in[23];
  const float* bh_ln_b  = (const float*)d_in[24];
  const float* sp_w     = (const float*)d_in[25];
  const float* sp_b     = (const float*)d_in[26];
  const int E = in_sizes[0];

  size_t off = 0;
  auto alloc = [&](size_t bytes) -> void* {
    size_t a = (off + 255) & ~(size_t)255;
    off = a + bytes;
    return (void*)((char*)d_ws + a);
  };
  int* counts = (int*)alloc(NB * sizeof(int));
  int* pos    = (int*)alloc((size_t)E * sizeof(int));
  unsigned short* wq_b    = (unsigned short*)alloc((size_t)256 * 256 * 2);
  unsigned short* wkv_b   = (unsigned short*)alloc((size_t)256 * 512 * 2);
  unsigned short* wo_b    = (unsigned short*)alloc((size_t)256 * 256 * 2);
  unsigned short* w1_b    = (unsigned short*)alloc((size_t)256 * 1024 * 2);
  unsigned short* w2_b    = (unsigned short*)alloc((size_t)1024 * 256 * 2);
  unsigned short* wqkvs_b = (unsigned short*)alloc((size_t)2 * 256 * 768 * 2);
  unsigned short* wos_b   = (unsigned short*)alloc((size_t)2 * 256 * 256 * 2);
  unsigned short* w1s_b   = (unsigned short*)alloc((size_t)2 * 256 * 1024 * 2);
  unsigned short* w2s_b   = (unsigned short*)alloc((size_t)2 * 1024 * 256 * 2);
  unsigned short* bhq_b   = (unsigned short*)alloc((size_t)2 * 256 * 2);
  unsigned short* bhwq_b  = (unsigned short*)alloc((size_t)256 * 256 * 2);
  unsigned short* bhwkv_b = (unsigned short*)alloc((size_t)256 * 512 * 2);
  unsigned short* spw_b   = (unsigned short*)alloc((size_t)256 * 64 * 2);
  float*          lat   = (float*)alloc((size_t)NB * NLAT * D_DIM * 4);
  unsigned short* xln   = (unsigned short*)alloc((size_t)NB * NLAT * D_DIM * 2);
  unsigned short* qc    = (unsigned short*)alloc((size_t)NB * NLAT * D_DIM * 2);
  unsigned short* atb   = (unsigned short*)alloc((size_t)NB * NLAT * D_DIM * 2);
  unsigned short* qkvb  = (unsigned short*)alloc((size_t)NB * NLAT * 768 * 2);
  unsigned short* h1    = (unsigned short*)alloc((size_t)NB * NLAT * 1024 * 2);
  unsigned short* kvbb  = (unsigned short*)alloc((size_t)NB * NLAT * 512 * 2);
  float*          bo    = (float*)alloc((size_t)NB * 2 * D_DIM * 4);
  unsigned short* meanl = (unsigned short*)alloc((size_t)NB * D_DIM * 2);
  unsigned short* bqq   = (unsigned short*)alloc((size_t)16 * D_DIM * 2);
  unsigned short* kvbuf = (unsigned short*)alloc((size_t)NB * EMAXD * 512 * 2);

  auto cvt = [&](const float* s, unsigned short* d, int n) {
    cvt_kernel<<<dim3((n + 255) / 256), dim3(256), 0, stream>>>(s, d, n);
  };
  auto gemm = [&](const unsigned short* X, int M, int K, int N, const unsigned short* W,
                  const float* bias, const float* resid, void* out, int flags) {
    gemm_kernel<<<dim3((M + 15) / 16, N / 64), dim3(128), 0, stream>>>(X, M, K, N, W, bias, resid, out, flags);
  };
  auto lnrm = [&](const float* X, unsigned short* Y, const float* g, const float* bt, int rows) {
    ln_kernel<<<dim3(rows), dim3(256), 0, stream>>>(X, Y, g, bt);
  };

  (void)hipMemsetAsync(counts, 0, NB * sizeof(int), stream);
  pos_kernel<<<dim3((E + 255) / 256), dim3(256), 0, stream>>>(bidx, counts, pos, E);

  cvt(Wq_c, wq_b, 256 * 256);     cvt(Wkv_c, wkv_b, 256 * 512);
  cvt(Wo_c, wo_b, 256 * 256);     cvt(W1_c, w1_b, 256 * 1024);
  cvt(W2_c, w2_b, 1024 * 256);    cvt(Wqkv_s, wqkvs_b, 2 * 256 * 768);
  cvt(Wo_s, wos_b, 2 * 256 * 256); cvt(W1_s, w1s_b, 2 * 256 * 1024);
  cvt(W2_s, w2s_b, 2 * 1024 * 256); cvt(bh_query, bhq_b, 2 * 256);
  cvt(bh_wq, bhwq_b, 256 * 256);  cvt(bh_wkv, bhwkv_b, 256 * 512);
  cvt(sp_w, spw_b, 256 * 64);

  lat_init_kernel<<<dim3((NB * NLAT * D_DIM) / 256), dim3(256), 0, stream>>>(lat_i, lat, NB * NLAT * D_DIM);

  const int ML = NB * NLAT;  // 1024 latent rows
  const float scale = 0.17677669529663687f;  // 1/sqrt(32)

  // --- cross attention: latents <- events ---
  lnrm(lat, xln, nullptr, nullptr, ML);
  gemm(xln, ML, 256, 256, wq_b, nullptr, nullptr, qc, GEMM_OUT_BF16);
  tok_kv_kernel<<<dim3((E + 15) / 16), dim3(256), 0, stream>>>(
      neuron_ids, time_bins, values, bidx, pos, nemb, temb, vemb, wkv_b, kvbuf, E);
  flash_kernel<<<dim3(1, NHEAD, NB), dim3(128), 0, stream>>>(
      qc, (long long)NLAT * 256, 256,
      kvbuf, (long long)EMAXD * 512, 512, 0, 256,
      counts, 0, atb, 0, (long long)NLAT * 256, 256, NLAT, scale);
  gemm(atb, ML, 256, 256, wo_b, nullptr, lat, lat, 0);
  lnrm(lat, xln, nullptr, nullptr, ML);
  gemm(xln, ML, 256, 1024, w1_b, nullptr, nullptr, h1, GEMM_OUT_BF16 | GEMM_GELU);
  gemm(h1, ML, 1024, 256, w2_b, nullptr, lat, lat, 0);

  // --- self-attention layers ---
  for (int i = 0; i < 2; ++i) {
    lnrm(lat, xln, nullptr, nullptr, ML);
    gemm(xln, ML, 256, 768, wqkvs_b + (size_t)i * 256 * 768, nullptr, nullptr, qkvb, GEMM_OUT_BF16);
    flash_kernel<<<dim3(1, NHEAD, NB), dim3(128), 0, stream>>>(
        qkvb, (long long)NLAT * 768, 768,
        qkvb, (long long)NLAT * 768, 768, 256, 512,
        nullptr, NLAT, atb, 0, (long long)NLAT * 256, 256, NLAT, scale);
    gemm(atb, ML, 256, 256, wos_b + (size_t)i * 256 * 256, nullptr, lat, lat, 0);
    lnrm(lat, xln, nullptr, nullptr, ML);
    gemm(xln, ML, 256, 1024, w1s_b + (size_t)i * 256 * 1024, nullptr, nullptr, h1, GEMM_OUT_BF16 | GEMM_GELU);
    gemm(h1, ML, 1024, 256, w2s_b + (size_t)i * 1024 * 256, nullptr, lat, lat, 0);
  }

  // --- behavior decoder ---
  lnrm(lat, xln, bh_ln_w, bh_ln_b, ML);
  gemm(bhq_b, 2, 256, 256, bhwq_b, nullptr, nullptr, bqq, GEMM_OUT_BF16);
  gemm(xln, ML, 256, 512, bhwkv_b, nullptr, nullptr, kvbb, GEMM_OUT_BF16);
  flash_kernel<<<dim3(1, NHEAD, NB), dim3(128), 0, stream>>>(
      bqq, 0, 256,
      kvbb, (long long)NLAT * 512, 512, 0, 256,
      nullptr, NLAT, bo, 1, (long long)2 * 256, 256, 2, scale);
  bdot_kernel<<<dim3(NB * 2), dim3(256), 0, stream>>>(bo, bh_wo, (float*)d_out);

  // --- spike head ---
  mean_kernel<<<dim3(NB), dim3(256), 0, stream>>>(lat, meanl);
  gemm(meanl, NB, 256, 64, spw_b, sp_b, nullptr, (float*)d_out + NB * 2, 0);
}
